// ConvSE3_56813827391796
// MI455X (gfx1250) — matrix operand order repaired and checked by reading, not compiled, not run on hardware
//
#include <hip/hip_runtime.h>
#include <hip/hip_bf16.h>

typedef __attribute__((ext_vector_type(16))) _Float16 v16h;
typedef __attribute__((ext_vector_type(8)))  _Float16 f16x8;
typedef __attribute__((ext_vector_type(8)))  float    v8f;

#define NPTS  512
#define JN    48
#define MDIM  16
#define MIDC  128
#define EDGES (NPTS*JN)      /* 24576 */
#define EPSLN 1e-5f

// ---------------------------------------------------------------------------
// Convert a row-major f32 [128 x Ncols] weight matrix into fp16 WMMA-B
// fragment layout: dst[((kk*NT + nt)*32 + lane)*16 + j] where
//   lanes 0-15 hold column (nt*16 + lane), K rows kk*32 + 0..7, 16..23  (j = 0..15)
//   lanes 16-31 hold column (nt*16 + lane-16), K rows kk*32 + 8..15, 24..31
// ---------------------------------------------------------------------------
__global__ void frag_prep(const float* __restrict__ src,
                          _Float16* __restrict__ dst, int Ncols) {
    int t = blockIdx.x * blockDim.x + threadIdx.x;
    int total = 128 * Ncols;
    if (t >= total) return;
    int j    = t & 15;
    int lane = (t >> 4) & 31;
    int rest = t >> 9;            // kk*NT + nt
    int NT   = Ncols >> 4;
    int nt   = rest % NT;
    int kk   = rest / NT;
    int k    = kk * 32 + ((lane < 16) ? 0 : 8) + (j & 7) + ((j & 8) ? 16 : 0);
    int col  = nt * 16 + (lane & 15);
    dst[t] = (_Float16)src[(size_t)k * Ncols + col];
}

// A-fragment loader: 16x32 fp16 A tile from LDS (row-major [16][128]).
// Lane l<16: row l, K = kbase+0..7 (elems 0-7) and kbase+16..23 (elems 8-15).
// Lane l>=16: row l-16, K = kbase+8..15 and kbase+24..31.
__device__ inline v16h load_a_frag(const _Float16* sh, int row, int kbase, bool hi) {
    const f16x8* p0 = (const f16x8*)(sh + row * 128 + kbase + (hi ? 8 : 0));
    const f16x8* p1 = (const f16x8*)(sh + row * 128 + kbase + 16 + (hi ? 8 : 0));
    f16x8 lo = *p0, hv = *p1;
    return __builtin_shufflevector(lo, hv, 0,1,2,3,4,5,6,7,8,9,10,11,12,13,14,15);
}

// ---------------------------------------------------------------------------
// Main fused kernel: one workgroup (8 wave32s) per (16-edge tile, pair).
// ---------------------------------------------------------------------------
__global__ __launch_bounds__(256)
void conv_main(const float* __restrict__ rel_dist, const int* __restrict__ nidx,
               const float* __restrict__ x0,  const float* __restrict__ x1,
               const float* __restrict__ basis00, const float* __restrict__ basis01,
               const float* __restrict__ basis10, const float* __restrict__ basis11,
               const float* __restrict__ w1, const float* __restrict__ b1,
               const float* __restrict__ g1, const float* __restrict__ be1,
               const _Float16* __restrict__ w2fr, const float* __restrict__ b2,
               const float* __restrict__ g2, const float* __restrict__ be2,
               const _Float16* __restrict__ w3fr0, const _Float16* __restrict__ w3fr1,
               const _Float16* __restrict__ w3fr2, const _Float16* __restrict__ w3fr3,
               const float* __restrict__ b3_0, const float* __restrict__ b3_1,
               const float* __restrict__ b3_2, const float* __restrict__ b3_3,
               float* __restrict__ edge_out)
{
    // 48 KB region: holds h1(4K)+h2(4K)+raw(8K) early, re-used as R[16][NT*16] late.
    __shared__ __align__(32) char uni[49152];
    __shared__ float sh_t[16][144];          // basis (x) gathered-x tensor
    __shared__ float sh_p1[16][8];           // LN partial sums
    __shared__ float sh_p2[16][8];           // LN partial sums of squares

    _Float16 (*sh_h1)[128] = (_Float16(*)[128])(uni);
    _Float16 (*sh_h2)[128] = (_Float16(*)[128])(uni + 4096);
    float    (*sh_raw)[128] = (float(*)[128])(uni + 8192);
    float    *sh_R          = (float*)(uni);

    const int p    = blockIdx.x & 3;        // pair index: 0:(0,0) 1:(0,1) 2:(1,0) 3:(1,1)
    const int tile = blockIdx.x >> 2;
    const int e0   = tile * 16;
    const int tid  = threadIdx.x;
    const int lane = tid & 31;
    const int wv   = tid >> 5;
    const bool hi  = (lane >= 16);

    // ---- phase 1: layer-1 pre-activations (threads 0-127, 8 segs x 16 edges)
    //               t-tensor (threads 128-255) ----
    if (tid < 128) {
        const int e  = tid >> 3;
        const int c0 = (tid & 7) * 16;
        const float d = rel_dist[e0 + e];
        const float* W1 = w1 + p * 128;  const float* B1 = b1 + p * 128;
        float s1 = 0.f, s2 = 0.f;
        #pragma unroll
        for (int c = c0; c < c0 + 16; c++) {
            float v = d * W1[c] + B1[c];
            sh_raw[e][c] = v; s1 += v; s2 += v * v;
        }
        sh_p1[e][tid & 7] = s1; sh_p2[e][tid & 7] = s2;
    } else {
        const int u = tid - 128;
        if (p == 0) {
            if (u < 16) {
                int e = e0 + u; int nb = nidx[e];
                float bv = basis00[e];
                for (int i = 0; i < 16; i++) sh_t[u][i] = bv * x0[nb * 16 + i];
            }
        } else if (p == 1) {
            if (u < 48) {
                int te = u / 3, a = u - te * 3;
                int e = e0 + te; int nb = nidx[e];
                float ba = basis01[e * 3 + a];
                for (int i = 0; i < 16; i++) sh_t[te][a * 16 + i] = ba * x0[nb * 16 + i];
            }
        } else if (p == 2) {
            if (u < 16) {
                int e = e0 + u; int nb = nidx[e];
                float bb0 = basis10[e * 3 + 0], bb1 = basis10[e * 3 + 1], bb2 = basis10[e * 3 + 2];
                for (int i = 0; i < 16; i++) {
                    const float* xv = x1 + (size_t)(nb * 16 + i) * 3;
                    sh_t[u][i] = bb0 * xv[0] + bb1 * xv[1] + bb2 * xv[2];
                }
            }
        } else {
            if (u < 48) {
                int te = u / 3, a = u - te * 3;
                int e = e0 + te; int nb = nidx[e];
                const float* bs = basis11 + (size_t)e * 27 + a * 9;  // idx = b*3+fi
                for (int i = 0; i < 16; i++) {
                    const float* xv = x1 + (size_t)(nb * 16 + i) * 3;
                    #pragma unroll
                    for (int fi = 0; fi < 3; fi++)
                        sh_t[te][a * 48 + i * 3 + fi] =
                            bs[0 * 3 + fi] * xv[0] + bs[1 * 3 + fi] * xv[1] + bs[2 * 3 + fi] * xv[2];
                }
            }
        }
    }
    __syncthreads();

    // ---- phase 1b: finish LN1+ReLU (threads 0-127; stats combined redundantly) ----
    if (tid < 128) {
        const int e  = tid >> 3;
        const int c0 = (tid & 7) * 16;
        float s1 = 0.f, s2 = 0.f;
        #pragma unroll
        for (int k = 0; k < 8; k++) { s1 += sh_p1[e][k]; s2 += sh_p2[e][k]; }
        float mu = s1 * (1.f / 128.f);
        float var = s2 * (1.f / 128.f) - mu * mu;
        float rs = rsqrtf(var + EPSLN);
        const float* G1 = g1 + p * 128;  const float* BE1 = be1 + p * 128;
        #pragma unroll
        for (int c = c0; c < c0 + 16; c++) {
            float h = (sh_raw[e][c] - mu) * rs * G1[c] + BE1[c];
            sh_h1[e][c] = (_Float16)fmaxf(h, 0.f);
        }
    }
    __syncthreads();

    // ---- phase 2: h2raw = h1 @ w2 + b2  (16x128 @ 128x128, 8 waves x 1 ntile) ----
    {
        const int row = lane & 15;
        v16h afr[4];
        #pragma unroll
        for (int kk = 0; kk < 4; kk++) afr[kk] = load_a_frag(&sh_h1[0][0], row, kk * 32, hi);

        const _Float16* wp = w2fr + (size_t)p * (128 * 128);
        const int nt = wv;                     // 8 waves cover 8 column tiles
        v8f acc = {};
        #pragma unroll
        for (int kk = 0; kk < 4; kk++) {
            v16h bfr = *(const v16h*)(wp + ((size_t)(kk * 8 + nt) * 32 + lane) * 16);
            acc = __builtin_amdgcn_wmma_f32_16x16x32_f16(
                      false, afr[kk], false, bfr, (short)0, acc, false, false);
        }
        const int col = nt * 16 + (lane & 15);
        const float bias = b2[p * 128 + col];
        #pragma unroll
        for (int r = 0; r < 8; r++) sh_raw[r + (hi ? 8 : 0)][col] = acc[r] + bias;
    }
    __syncthreads();

    // ---- phase 3: h2 = relu(ln(h2raw)) -> fp16, parallel over 128 threads ----
    if (tid < 128) {
        const int e  = tid >> 3;
        const int c0 = (tid & 7) * 16;
        float s1 = 0.f, s2 = 0.f;
        #pragma unroll
        for (int c = c0; c < c0 + 16; c++) { float v = sh_raw[e][c]; s1 += v; s2 += v * v; }
        sh_p1[e][tid & 7] = s1; sh_p2[e][tid & 7] = s2;
    }
    __syncthreads();
    if (tid < 128) {
        const int e  = tid >> 3;
        const int c0 = (tid & 7) * 16;
        float s1 = 0.f, s2 = 0.f;
        #pragma unroll
        for (int k = 0; k < 8; k++) { s1 += sh_p1[e][k]; s2 += sh_p2[e][k]; }
        float mu = s1 * (1.f / 128.f);
        float var = s2 * (1.f / 128.f) - mu * mu;
        float rs = rsqrtf(var + EPSLN);
        const float* G2 = g2 + p * 128;  const float* BE2 = be2 + p * 128;
        #pragma unroll
        for (int c = c0; c < c0 + 16; c++) {
            float h = (sh_raw[e][c] - mu) * rs * G2[c] + BE2[c];
            sh_h2[e][c] = (_Float16)fmaxf(h, 0.f);
        }
    }
    __syncthreads();

    // ---- phase 4: R = h2 @ w3 + b3 -> LDS (R aliases h1/h2/raw; barrier below) ----
    {
        const int row = lane & 15;
        v16h afr[4];
        #pragma unroll
        for (int kk = 0; kk < 4; kk++) afr[kk] = load_a_frag(&sh_h2[0][0], row, kk * 32, hi);
        __syncthreads();   // everyone has A-fragments in registers; uni[] is now free for R

        const _Float16* wfr = (p == 0) ? w3fr0 : (p == 1) ? w3fr1 : (p == 2) ? w3fr2 : w3fr3;
        const float*    b3p = (p == 0) ? b3_0  : (p == 1) ? b3_1  : (p == 2) ? b3_2  : b3_3;
        const int NT      = (p == 3) ? 48 : 16;
        const int rstride = NT * 16;          // 768 or 256 cols per edge
        const int eoff    = hi ? 8 : 0;

        for (int nt = wv; nt < NT; nt += 8) {
            v8f acc = {};
            #pragma unroll
            for (int kk = 0; kk < 4; kk++) {
                v16h bfr = *(const v16h*)(wfr + ((size_t)(kk * NT + nt) * 32 + lane) * 16);
                acc = __builtin_amdgcn_wmma_f32_16x16x32_f16(
                          false, afr[kk], false, bfr, (short)0, acc, false, false);
            }
            const int col  = nt * 16 + (lane & 15);
            const float bias = b3p[col];
            #pragma unroll
            for (int r = 0; r < 8; r++)
                sh_R[(r + eoff) * rstride + col] = acc[r] + bias;
        }
    }
    __syncthreads();

    // ---- phase 5: deterministic contraction; thread (e,o) owns one output group ----
    {
        const int e = tid >> 4;               // 0..15
        const int o = tid & 15;               // 0..15
        const int choff = (p == 0) ? 0 : (p == 1) ? 16 : (p == 2) ? 64 : 80;
        float* eop = edge_out + (size_t)(e0 + e) * 128 + choff;

        if (p == 3) {                         // acc[a] = sum_{i,fi} R[(o*16+i)*3+fi] * t[a*48+i*3+fi]
            const float* Re = sh_R + e * 768 + o * 48;
            float a0 = 0.f, a1 = 0.f, a2 = 0.f;
            #pragma unroll
            for (int i = 0; i < 16; i++) {
                #pragma unroll
                for (int fi = 0; fi < 3; fi++) {
                    float rv = Re[i * 3 + fi];
                    a0 += rv * sh_t[e][0 * 48 + i * 3 + fi];
                    a1 += rv * sh_t[e][1 * 48 + i * 3 + fi];
                    a2 += rv * sh_t[e][2 * 48 + i * 3 + fi];
                }
            }
            eop[o * 3 + 0] = a0; eop[o * 3 + 1] = a1; eop[o * 3 + 2] = a2;
        } else if (p == 1) {                  // acc[a] = sum_i R[o*16+i] * t[a*16+i]
            const float* Re = sh_R + e * 256 + o * 16;
            float a0 = 0.f, a1 = 0.f, a2 = 0.f;
            #pragma unroll
            for (int i = 0; i < 16; i++) {
                float rv = Re[i];
                a0 += rv * sh_t[e][i];
                a1 += rv * sh_t[e][16 + i];
                a2 += rv * sh_t[e][32 + i];
            }
            eop[o * 3 + 0] = a0; eop[o * 3 + 1] = a1; eop[o * 3 + 2] = a2;
        } else {                              // acc = sum_i R[o*16+i] * t[i]
            const float* Re = sh_R + e * 256 + o * 16;
            float a0 = 0.f;
            #pragma unroll
            for (int i = 0; i < 16; i++) a0 += Re[i] * sh_t[e][i];
            eop[o] = a0;
        }
    }
}

// ---------------------------------------------------------------------------
// Masked mean over neighbors + self-interaction + [B,N,M,4] packing.
// ---------------------------------------------------------------------------
__global__ void combine(const float* __restrict__ eo, const int* __restrict__ mask,
                        const float* __restrict__ x0, const float* __restrict__ x1,
                        const float* __restrict__ ks0, const float* __restrict__ ks1,
                        float* __restrict__ out)
{
    const int n = blockIdx.x;
    const int c = threadIdx.x;       // 64 output channels per point
    float denom = 0.f;
    for (int j = 0; j < 48; j++) denom += mask[n * 48 + j] ? 1.f : 0.f;

    if (c < 16) {                    // degree-0 output, m = c
        const int m = c;
        float s = 0.f;
        for (int j = 0; j < 48; j++) {
            if (mask[n * 48 + j]) {
                const float* ep = eo + (size_t)(n * 48 + j) * 128;
                s += ep[m] + ep[64 + m];
            }
        }
        float self = 0.f;
        for (int i = 0; i < 16; i++) self += ks0[m * 16 + i] * x0[n * 16 + i];
        out[(size_t)(n * 16 + m) * 4 + 0] = s / denom + self;
    } else {                         // degree-1 output, m = (c-16)/3, component dd
        const int cc = c - 16;
        const int m = cc / 3; const int dd = cc - m * 3;
        float s = 0.f;
        for (int j = 0; j < 48; j++) {
            if (mask[n * 48 + j]) {
                const float* ep = eo + (size_t)(n * 48 + j) * 128;
                s += ep[16 + m * 3 + dd] + ep[80 + m * 3 + dd];
            }
        }
        float self = 0.f;
        for (int i = 0; i < 16; i++) self += ks1[m * 16 + i] * x1[(size_t)(n * 16 + i) * 3 + dd];
        out[(size_t)(n * 16 + m) * 4 + 1 + dd] = s / denom + self;
    }
}

// ---------------------------------------------------------------------------
extern "C" void kernel_launch(void* const* d_in, const int* in_sizes, int n_in,
                              void* d_out, int out_size, void* d_ws, size_t ws_size,
                              hipStream_t stream)
{
    (void)in_sizes; (void)n_in; (void)out_size; (void)ws_size;
    const float* x0   = (const float*)d_in[0];
    const float* x1   = (const float*)d_in[1];
    const float* rel  = (const float*)d_in[2];
    const int*   nidx = (const int*)d_in[3];
    const int*   mask = (const int*)d_in[4];
    const float* b00  = (const float*)d_in[5];
    const float* b01  = (const float*)d_in[6];
    const float* b10  = (const float*)d_in[7];
    const float* b11  = (const float*)d_in[8];
    const float* w1   = (const float*)d_in[9];
    const float* b1   = (const float*)d_in[10];
    const float* g1   = (const float*)d_in[11];
    const float* be1  = (const float*)d_in[12];
    const float* w2   = (const float*)d_in[13];
    const float* b2   = (const float*)d_in[14];
    const float* g2   = (const float*)d_in[15];
    const float* be2  = (const float*)d_in[16];
    const float* w3_00 = (const float*)d_in[17];
    const float* b3_00 = (const float*)d_in[18];
    const float* w3_01 = (const float*)d_in[19];
    const float* b3_01 = (const float*)d_in[20];
    const float* w3_10 = (const float*)d_in[21];
    const float* b3_10 = (const float*)d_in[22];
    const float* w3_11 = (const float*)d_in[23];
    const float* b3_11 = (const float*)d_in[24];
    const float* ks0   = (const float*)d_in[25];
    const float* ks1   = (const float*)d_in[26];

    char* ws = (char*)d_ws;
    _Float16* w2fr  = (_Float16*)(ws + 0);          // 4*128*128 fp16 = 131072 B
    _Float16* w3fr0 = (_Float16*)(ws + 131072);     // 128*256 fp16  =  65536 B
    _Float16* w3fr1 = (_Float16*)(ws + 196608);
    _Float16* w3fr2 = (_Float16*)(ws + 262144);
    _Float16* w3fr3 = (_Float16*)(ws + 327680);     // 128*768 fp16  = 196608 B
    float*    eo    = (float*)(ws + 524288);        // EDGES*128 f32 = 12.6 MB

    // weight -> fp16 fragment layout
    for (int p = 0; p < 4; p++)
        frag_prep<<<(128 * 128 + 255) / 256, 256, 0, stream>>>(w2 + (size_t)p * 128 * 128,
                                                               w2fr + (size_t)p * 128 * 128, 128);
    frag_prep<<<(128 * 256 + 255) / 256, 256, 0, stream>>>(w3_00, w3fr0, 256);
    frag_prep<<<(128 * 256 + 255) / 256, 256, 0, stream>>>(w3_01, w3fr1, 256);
    frag_prep<<<(128 * 256 + 255) / 256, 256, 0, stream>>>(w3_10, w3fr2, 256);
    frag_prep<<<(128 * 768 + 255) / 256, 256, 0, stream>>>(w3_11, w3fr3, 768);

    // fused radial-MLP (WMMA) + kernel contraction, per (16-edge tile, pair)
    conv_main<<<(EDGES / 16) * 4, 256, 0, stream>>>(
        rel, nidx, x0, x1, b00, b01, b10, b11,
        w1, b1, g1, be1, w2fr, b2, g2, be2,
        w3fr0, w3fr1, w3fr2, w3fr3, b3_00, b3_01, b3_10, b3_11, eo);

    // masked mean + self-interaction + packing
    combine<<<NPTS, 64, 0, stream>>>(eo, mask, x0, x1, ks0, ks1, (float*)d_out);
}
